// LSTM_32238024524121
// MI455X (gfx1250) — compile-verified, matrix operand order checked by reading, not run on hardware
//
#include <hip/hip_runtime.h>
#include <cstdint>

// ---------------------------------------------------------------------------
// Seq2Seq LSTM encoder/decoder for MI455X (gfx1250), wave32 + WMMA bf16.
//
//   1. convert all weights f32 -> bf16 (once per call)
//   2. embed src (flipped) -> seq0 bf16 ; embed tgt -> inp0[:, :1024]
//   3. per enc layer: big GEMM (x @ W_ih^T + b) for both dirs, then 64
//      per-timestep WMMA recurrence kernels (gate-strip ownership)
//   4. decoder init from encoder final states; context -> inp0[:, 1024:]
//   5. big GEMM P0 = inp0 @ W_ih0^T + db0 ; 63 x (cell0, cell1) step kernels
//   6. logits GEMM h1_all @ W_out^T + b_out -> d_out (row remap t,b -> b,t)
//
// Big GEMM: 128Mx128N block tile, A panel staged in LDS via CDNA5
// async-to-LDS loads (ASYNCcnt) when available; W fragments from L2.
// ---------------------------------------------------------------------------

typedef __attribute__((ext_vector_type(16))) __bf16 v16bf;
typedef __attribute__((ext_vector_type(8)))  float  v8f;
typedef int v4i_ __attribute__((vector_size(16)));

#if defined(__has_builtin)
#if __has_builtin(__builtin_amdgcn_global_load_async_to_lds_b128)
#define ASYNC_LDS 1
#endif
#endif

static __device__ inline v8f vzero8() {
  v8f v = {0.f, 0.f, 0.f, 0.f, 0.f, 0.f, 0.f, 0.f};
  return v;
}

static __device__ inline uint16_t f2bf(float x) {
  uint32_t u = __float_as_uint(x);
  u += 0x7fffu + ((u >> 16) & 1u);           // round-to-nearest-even
  return (uint16_t)(u >> 16);
}

static __device__ inline float sigm(float x) {
  return 1.0f / (1.0f + __expf(-x));
}

// D = A(16x32) * B(32x16) + C, bf16 in / f32 accum
static __device__ inline v8f wmma_bf16(v16bf a, v16bf b, v8f c) {
  return __builtin_amdgcn_wmma_f32_16x16x32_bf16(
      /*neg_a=*/false, a, /*neg_b=*/false, b,
      /*c_mod=*/(short)0, c, /*reuse_a=*/false, /*reuse_b=*/false);
}

// Load one 16x32 bf16 fragment (A layout; also used for B via W[N,K] rows).
// Per ISA 16-bit A layout: lanes 0-15 -> row=lane,    K chunks {0..7,16..23}
//                          lanes16-31 -> row=lane-16, K chunks {8..15,24..31}
union FragU { uint4 q[2]; v16bf v; };
static __device__ inline v16bf load_frag(const uint16_t* __restrict__ p, int ld) {
  int lane = threadIdx.x & 31;
  int r    = lane & 15;
  int ko   = (lane & 16) ? 8 : 0;
  const uint16_t* q = p + (long)r * ld + ko;
  FragU u;
  u.q[0] = *(const uint4*)(q);
  u.q[1] = *(const uint4*)(q + 16);
  return u.v;
}

// 16-byte copy global -> LDS (async-to-LDS path when the toolchain has it)
static __device__ inline void cp16_g2l(const uint16_t* __restrict__ g,
                                       uint16_t* __restrict__ l) {
#ifdef ASYNC_LDS
  __builtin_amdgcn_global_load_async_to_lds_b128(
      (__attribute__((address_space(1))) v4i_*)g,
      (__attribute__((address_space(3))) v4i_*)l, 0, 0);
#else
  *(uint4*)l = *(const uint4*)g;
#endif
}

static __device__ inline void cp_wait() {
#ifdef ASYNC_LDS
#if __has_builtin(__builtin_amdgcn_s_wait_asynccnt)
  __builtin_amdgcn_s_wait_asynccnt(0);
#else
  asm volatile("s_wait_asynccnt 0x0" ::: "memory");
#endif
#endif
}

// ---------------------------------------------------------------------------
// elementwise helpers
// ---------------------------------------------------------------------------
__global__ void cvt_bf16_k(const float* __restrict__ s, uint16_t* __restrict__ d, long n) {
  long i = (long)blockIdx.x * blockDim.x + threadIdx.x;
  long st = (long)gridDim.x * blockDim.x;
  for (; i < n; i += st) d[i] = f2bf(s[i]);
}

__global__ void add2_k(const float* __restrict__ a, const float* __restrict__ b,
                       float* __restrict__ d, int n) {
  int i = blockIdx.x * blockDim.x + threadIdx.x;
  if (i < n) d[i] = a[i] + b[i];
}

// seq0[s][b][e] = bf16(src_emb[x[b][63-s]][e])   (flip on src axis)
__global__ void embed_src_k(const int* __restrict__ x, const float* __restrict__ emb,
                            uint16_t* __restrict__ seq) {
  int s = blockIdx.x, b = blockIdx.y;
  int tok = x[b * 64 + (63 - s)];
  const float* e = emb + (long)tok * 1024;
  uint16_t* o = seq + ((long)s * 32 + b) * 1024;
  for (int i = threadIdx.x; i < 1024; i += blockDim.x) o[i] = f2bf(e[i]);
}

// inp0[t][b][0:1024] = bf16(tgt_emb[y[b][t]])    (teacher forcing, y[:, :-1])
__global__ void embed_tgt_k(const int* __restrict__ y, const float* __restrict__ emb,
                            uint16_t* __restrict__ inp0) {
  int t = blockIdx.x, b = blockIdx.y;
  int tok = y[b * 64 + t];
  const float* e = emb + (long)tok * 1024;
  uint16_t* o = inp0 + ((long)t * 32 + b) * 2048;
  for (int i = threadIdx.x; i < 1024; i += blockDim.x) o[i] = f2bf(e[i]);
}

// inp0[t][b][1024:2048] = context[b][:]  (context = enc layer-1 final hidden)
__global__ void ctx_fill_k(const uint16_t* __restrict__ ctx, uint16_t* __restrict__ inp0) {
  int t = blockIdx.x, b = blockIdx.y;
  const uint16_t* s = ctx + (long)b * 1024;
  uint16_t* o = inp0 + ((long)t * 32 + b) * 2048 + 1024;
  for (int i = threadIdx.x; i < 1024; i += blockDim.x) o[i] = s[i];
}

// decoder state init from encoder finals
__global__ void dec_init_k(const uint16_t* __restrict__ hfin, const float* __restrict__ cfin,
                           uint16_t* __restrict__ h0p, uint16_t* __restrict__ h1p,
                           float* __restrict__ c0, float* __restrict__ c1) {
  int i = blockIdx.x * blockDim.x + threadIdx.x;
  if (i < 32 * 1024) {
    h0p[i] = hfin[i];
    h1p[i] = hfin[32 * 1024 + i];
    c0[i]  = cfin[i];
    c1[i]  = cfin[32 * 1024 + i];
  }
}

// ---------------------------------------------------------------------------
// Big GEMM: C[M,N] = A[M,K](bf16) * W[N,K]^T(bf16) + bias[N]
// block = 256 threads (8 waves, 2Mx4N). block tile 128M x 128N; wave tile
// 64M x 32N (4x2 accumulator tiles). A panel chunk (128x64) staged in LDS
// (async-to-LDS), W fragments streamed from global/L2.
// A buffer must have ceil(M/128)*128 rows (zero-padded); stores guarded m<M.
// REMAP: store row m -> out[b = m&31][t = m>>5]  (logits -> d_out [B,T-1,V])
// ---------------------------------------------------------------------------
#define GEMM_LDA 72

template <bool REMAP>
__global__ void gemm_k(const uint16_t* __restrict__ A, const uint16_t* __restrict__ W,
                       const float* __restrict__ bias, float* __restrict__ C,
                       int M, int N, int K, long rem_bstride, long rem_tstride) {
  __shared__ uint16_t As[128 * GEMM_LDA];
  int tid  = threadIdx.x;
  int wave = tid >> 5;
  int lane = tid & 31;
  int wm   = wave >> 2;                       // 0..1
  int wn   = wave & 3;                        // 0..3
  int m0   = blockIdx.y * 128;
  int n0   = blockIdx.x * 128 + wn * 32;

  v8f acc[4][2];
#pragma unroll
  for (int mt = 0; mt < 4; ++mt) {
    acc[mt][0] = vzero8();
    acc[mt][1] = vzero8();
  }

  const uint16_t* W0 = W + (long)n0 * K;
  const uint16_t* W1 = W + (long)(n0 + 16) * K;

  int srow = tid >> 1;                        // 0..127
  int scol = (tid & 1) * 32;                  // 0 or 32
  const uint16_t* Ag = A + (long)(m0 + srow) * K + scol;
  uint16_t* Al = &As[srow * GEMM_LDA + scol];

  for (int k0 = 0; k0 < K; k0 += 64) {
    __syncthreads();                          // prior chunk's reads done
#pragma unroll
    for (int i = 0; i < 4; ++i) cp16_g2l(Ag + k0 + i * 8, Al + i * 8);
    cp_wait();                                // this wave's copies landed
    __syncthreads();                          // everyone's copies landed
#pragma unroll
    for (int ki = 0; ki < 2; ++ki) {
      int kk = ki * 32;
      v16bf b0 = load_frag(W0 + k0 + kk, K);
      v16bf b1 = load_frag(W1 + k0 + kk, K);
#pragma unroll
      for (int mt = 0; mt < 4; ++mt) {
        v16bf a = load_frag(&As[(wm * 64 + mt * 16) * GEMM_LDA + kk], GEMM_LDA);
        acc[mt][0] = wmma_bf16(a, b0, acc[mt][0]);
        acc[mt][1] = wmma_bf16(a, b1, acc[mt][1]);
      }
    }
  }

  int n  = lane & 15;
  int mo = (lane & 16) ? 8 : 0;
  float bv0 = bias ? bias[n0 + n] : 0.f;
  float bv1 = bias ? bias[n0 + 16 + n] : 0.f;

#pragma unroll
  for (int mt = 0; mt < 4; ++mt) {
#pragma unroll
    for (int r = 0; r < 8; ++r) {
      int m = m0 + wm * 64 + mt * 16 + mo + r;
      if (m < M) {
        if (!REMAP) {
          C[(long)m * N + n0 + n]      = acc[mt][0][r] + bv0;
          C[(long)m * N + n0 + 16 + n] = acc[mt][1][r] + bv1;
        } else {
          long o = (long)(m & 31) * rem_bstride + (long)(m >> 5) * rem_tstride;
          C[o + n0 + n]      = acc[mt][0][r] + bv0;
          C[o + n0 + 16 + n] = acc[mt][1][r] + bv1;
        }
      }
    }
  }
}

// ---------------------------------------------------------------------------
// Encoder recurrence, one timestep, both directions (blockIdx.y = dir).
// grid (4, 2), block 256. Each wave owns a 16-wide hidden strip and computes
// the i,f,g,o gate tiles (4 gates x 2 M-tiles) via WMMA over K=512, then the
// LSTM elementwise update in registers. h double-buffered bf16; c f32 (owned).
// ---------------------------------------------------------------------------
__global__ void enc_step_k(const uint16_t* __restrict__ Whh,   // [2][2048*512] (layer base)
                           const float* __restrict__ gp,       // [2][2048*2048] precomp gates
                           uint16_t* __restrict__ hpool,       // [2 dir][2 buf][32*512]
                           float* __restrict__ cpool,          // [2 dir][32*512]
                           uint16_t* __restrict__ seq_out,     // [64][32][1024] or null
                           uint16_t* __restrict__ hfin,        // [32][1024] (t==63 only)
                           float* __restrict__ cfin,           // [32][1024] (t==63 only)
                           int t) {
  int dir  = blockIdx.y;
  int wave = threadIdx.x >> 5;
  int lane = threadIdx.x & 31;
  int j0   = (blockIdx.x * 8 + wave) * 16;           // hidden strip [j0, j0+16)
  int trow = dir ? (63 - t) : t;                     // reversed seq for bwd
  int n    = lane & 15;
  int mo   = (lane & 16) ? 8 : 0;

  const uint16_t* Wd    = Whh + (long)dir * 2048 * 512;
  const float*    gpd   = gp  + (long)dir * 2048 * 2048;
  const uint16_t* hprev = hpool + ((long)dir * 2 + (t & 1))       * (32 * 512);
  uint16_t*       hnext = hpool + ((long)dir * 2 + ((t + 1) & 1)) * (32 * 512);
  float*          c     = cpool + (long)dir * (32 * 512);

  v8f acc[4][2];
#pragma unroll
  for (int g = 0; g < 4; ++g)
#pragma unroll
    for (int mt = 0; mt < 2; ++mt) {
      const float* src = gpd + (long)(trow * 32 + mt * 16 + mo) * 2048 + g * 512 + j0 + n;
      v8f a;
#pragma unroll
      for (int r = 0; r < 8; ++r) a[r] = src[(long)r * 2048];
      acc[g][mt] = a;
    }

  for (int kk = 0; kk < 512; kk += 32) {
    v16bf a0 = load_frag(hprev + kk, 512);
    v16bf a1 = load_frag(hprev + 16 * 512 + kk, 512);
#pragma unroll
    for (int g = 0; g < 4; ++g) {
      v16bf b = load_frag(Wd + (long)(g * 512 + j0) * 512 + kk, 512);
      acc[g][0] = wmma_bf16(a0, b, acc[g][0]);
      acc[g][1] = wmma_bf16(a1, b, acc[g][1]);
    }
  }

  int j = j0 + n;
#pragma unroll
  for (int mt = 0; mt < 2; ++mt)
#pragma unroll
    for (int r = 0; r < 8; ++r) {
      int m = mt * 16 + mo + r;                      // batch row
      float iv = sigm(acc[0][mt][r]);
      float fv = sigm(acc[1][mt][r]);
      float gv = tanhf(acc[2][mt][r]);
      float ov = sigm(acc[3][mt][r]);
      long ci  = (long)m * 512 + j;
      float c2 = fv * c[ci] + iv * gv;
      c[ci] = c2;
      float h2 = ov * tanhf(c2);
      uint16_t hb = f2bf(h2);
      hnext[ci] = hb;
      if (seq_out) seq_out[((long)trow * 32 + m) * 1024 + dir * 512 + j] = hb;
      if (t == 63) {
        hfin[(long)m * 1024 + dir * 512 + j] = hb;
        cfin[(long)m * 1024 + dir * 512 + j] = c2;
      }
    }
}

// ---------------------------------------------------------------------------
// Decoder cell 0, one timestep. grid (8), block 256 (64 hidden strips).
// gates0 = P0[t] + h0 @ W_hh0^T, LSTM update of (h0, c0).
// ---------------------------------------------------------------------------
__global__ void dec_cell0_k(const uint16_t* __restrict__ Whh0,  // [4096][1024]
                            const float* __restrict__ P0,       // [63*32][4096]
                            uint16_t* __restrict__ h0pool,      // [2][32*1024]
                            float* __restrict__ c0,             // [32*1024]
                            int t) {
  int wave = threadIdx.x >> 5;
  int lane = threadIdx.x & 31;
  int j0   = (blockIdx.x * 8 + wave) * 16;
  int n    = lane & 15;
  int mo   = (lane & 16) ? 8 : 0;

  const uint16_t* hprev = h0pool + (long)(t & 1) * (32 * 1024);
  uint16_t*       hnext = h0pool + (long)((t + 1) & 1) * (32 * 1024);

  v8f acc[4][2];
#pragma unroll
  for (int g = 0; g < 4; ++g)
#pragma unroll
    for (int mt = 0; mt < 2; ++mt) {
      const float* src = P0 + (long)(t * 32 + mt * 16 + mo) * 4096 + g * 1024 + j0 + n;
      v8f a;
#pragma unroll
      for (int r = 0; r < 8; ++r) a[r] = src[(long)r * 4096];
      acc[g][mt] = a;
    }

  for (int kk = 0; kk < 1024; kk += 32) {
    v16bf a0 = load_frag(hprev + kk, 1024);
    v16bf a1 = load_frag(hprev + 16 * 1024 + kk, 1024);
#pragma unroll
    for (int g = 0; g < 4; ++g) {
      v16bf b = load_frag(Whh0 + (long)(g * 1024 + j0) * 1024 + kk, 1024);
      acc[g][0] = wmma_bf16(a0, b, acc[g][0]);
      acc[g][1] = wmma_bf16(a1, b, acc[g][1]);
    }
  }

  int j = j0 + n;
#pragma unroll
  for (int mt = 0; mt < 2; ++mt)
#pragma unroll
    for (int r = 0; r < 8; ++r) {
      int m = mt * 16 + mo + r;
      float iv = sigm(acc[0][mt][r]);
      float fv = sigm(acc[1][mt][r]);
      float gv = tanhf(acc[2][mt][r]);
      float ov = sigm(acc[3][mt][r]);
      long ci  = (long)m * 1024 + j;
      float c2 = fv * c0[ci] + iv * gv;
      c0[ci] = c2;
      hnext[ci] = f2bf(ov * tanhf(c2));
    }
}

// ---------------------------------------------------------------------------
// Decoder cell 1, one timestep. gates1 = h0 @ W_ih1^T + h1 @ W_hh1^T + db1.
// Writes (h1, c1) state and h1_all[t] for the final logits GEMM.
// ---------------------------------------------------------------------------
__global__ void dec_cell1_k(const uint16_t* __restrict__ Wih1,  // [4096][1024]
                            const uint16_t* __restrict__ Whh1,  // [4096][1024]
                            const float* __restrict__ db1,      // [4096]
                            const uint16_t* __restrict__ h0cur, // [32*1024] (this step's h0)
                            uint16_t* __restrict__ h1pool,      // [2][32*1024]
                            float* __restrict__ c1,             // [32*1024]
                            uint16_t* __restrict__ h1all,       // [63*32][1024] (padded)
                            int t) {
  int wave = threadIdx.x >> 5;
  int lane = threadIdx.x & 31;
  int j0   = (blockIdx.x * 8 + wave) * 16;
  int n    = lane & 15;
  int mo   = (lane & 16) ? 8 : 0;

  const uint16_t* hprev = h1pool + (long)(t & 1) * (32 * 1024);
  uint16_t*       hnext = h1pool + (long)((t + 1) & 1) * (32 * 1024);

  v8f acc[4][2];
#pragma unroll
  for (int g = 0; g < 4; ++g) {
    float bv = db1[g * 1024 + j0 + n];
    v8f a;
#pragma unroll
    for (int r = 0; r < 8; ++r) a[r] = bv;
    acc[g][0] = a;
    acc[g][1] = a;
  }

  for (int kk = 0; kk < 1024; kk += 32) {          // h0 @ W_ih1^T
    v16bf a0 = load_frag(h0cur + kk, 1024);
    v16bf a1 = load_frag(h0cur + 16 * 1024 + kk, 1024);
#pragma unroll
    for (int g = 0; g < 4; ++g) {
      v16bf b = load_frag(Wih1 + (long)(g * 1024 + j0) * 1024 + kk, 1024);
      acc[g][0] = wmma_bf16(a0, b, acc[g][0]);
      acc[g][1] = wmma_bf16(a1, b, acc[g][1]);
    }
  }
  for (int kk = 0; kk < 1024; kk += 32) {          // h1 @ W_hh1^T
    v16bf a0 = load_frag(hprev + kk, 1024);
    v16bf a1 = load_frag(hprev + 16 * 1024 + kk, 1024);
#pragma unroll
    for (int g = 0; g < 4; ++g) {
      v16bf b = load_frag(Whh1 + (long)(g * 1024 + j0) * 1024 + kk, 1024);
      acc[g][0] = wmma_bf16(a0, b, acc[g][0]);
      acc[g][1] = wmma_bf16(a1, b, acc[g][1]);
    }
  }

  int j = j0 + n;
#pragma unroll
  for (int mt = 0; mt < 2; ++mt)
#pragma unroll
    for (int r = 0; r < 8; ++r) {
      int m = mt * 16 + mo + r;
      float iv = sigm(acc[0][mt][r]);
      float fv = sigm(acc[1][mt][r]);
      float gv = tanhf(acc[2][mt][r]);
      float ov = sigm(acc[3][mt][r]);
      long ci  = (long)m * 1024 + j;
      float c2 = fv * c1[ci] + iv * gv;
      c1[ci] = c2;
      uint16_t hb = f2bf(ov * tanhf(c2));
      hnext[ci] = hb;
      h1all[((long)t * 32 + m) * 1024 + j] = hb;
    }
}

// ---------------------------------------------------------------------------
// host orchestration
// ---------------------------------------------------------------------------
extern "C" void kernel_launch(void* const* d_in, const int* in_sizes, int n_in,
                              void* d_out, int out_size, void* d_ws, size_t ws_size,
                              hipStream_t stream) {
  (void)in_sizes; (void)n_in; (void)out_size; (void)ws_size;

  const int*   x        = (const int*)d_in[0];
  const int*   y        = (const int*)d_in[1];
  const float* src_emb  = (const float*)d_in[2];
  const float* tgt_emb  = (const float*)d_in[3];
  const float* eWih     = (const float*)d_in[4];
  const float* eWhh     = (const float*)d_in[5];
  const float* eBih     = (const float*)d_in[6];
  const float* eBhh     = (const float*)d_in[7];
  const float* dWih0    = (const float*)d_in[8];
  const float* dWih1    = (const float*)d_in[9];
  const float* dWhh     = (const float*)d_in[10];
  const float* dBih     = (const float*)d_in[11];
  const float* dBhh     = (const float*)d_in[12];
  const float* Wout     = (const float*)d_in[13];
  const float* bout     = (const float*)d_in[14];
  float*       out      = (float*)d_out;

  char* ws = (char*)d_ws;
  size_t off = 0;
  auto alloc = [&](size_t bytes) -> void* {
    off = (off + 255) & ~(size_t)255;
    void* p = ws + off;
    off += bytes;
    return p;
  };

  uint16_t* eWih_b  = (uint16_t*)alloc((size_t)2 * 2 * 2048 * 1024 * 2);
  uint16_t* eWhh_b  = (uint16_t*)alloc((size_t)2 * 2 * 2048 * 512 * 2);
  uint16_t* dWih0_b = (uint16_t*)alloc((size_t)4096 * 2048 * 2);
  uint16_t* dWih1_b = (uint16_t*)alloc((size_t)4096 * 1024 * 2);
  uint16_t* dWhh_b  = (uint16_t*)alloc((size_t)2 * 4096 * 1024 * 2);
  uint16_t* Wout_b  = (uint16_t*)alloc((size_t)32000 * 1024 * 2);
  float*    bcomb   = (float*)alloc((size_t)2 * 2 * 2048 * 4);
  float*    dbcomb  = (float*)alloc((size_t)2 * 4096 * 4);
  uint16_t* seq0    = (uint16_t*)alloc((size_t)64 * 32 * 1024 * 2);
  uint16_t* seq1    = (uint16_t*)alloc((size_t)64 * 32 * 1024 * 2);
  float*    gp      = (float*)alloc((size_t)2 * 2048 * 2048 * 4);
  uint16_t* ehpool  = (uint16_t*)alloc((size_t)2 * 2 * 32 * 512 * 2);
  float*    ecpool  = (float*)alloc((size_t)2 * 32 * 512 * 4);
  uint16_t* hfin    = (uint16_t*)alloc((size_t)2 * 32 * 1024 * 2);
  float*    cfin    = (float*)alloc((size_t)2 * 32 * 1024 * 4);
  uint16_t* inp0    = (uint16_t*)alloc((size_t)2048 * 2048 * 2);   // 2048 rows (padded)
  float*    P0      = (float*)alloc((size_t)2016 * 4096 * 4);
  uint16_t* h0pool  = (uint16_t*)alloc((size_t)2 * 32 * 1024 * 2);
  uint16_t* h1pool  = (uint16_t*)alloc((size_t)2 * 32 * 1024 * 2);
  float*    c0      = (float*)alloc((size_t)32 * 1024 * 4);
  float*    c1      = (float*)alloc((size_t)32 * 1024 * 4);
  uint16_t* h1all   = (uint16_t*)alloc((size_t)2048 * 1024 * 2);   // 2048 rows (padded)

  // 0. zero the M-padding rows (rows 2016..2047) of padded A buffers
  (void)hipMemsetAsync(inp0 + (size_t)2016 * 2048, 0, (size_t)32 * 2048 * 2, stream);
  (void)hipMemsetAsync(h1all + (size_t)2016 * 1024, 0, (size_t)32 * 1024 * 2, stream);

  // 1. weight conversion f32 -> bf16
  cvt_bf16_k<<<2048, 256, 0, stream>>>(eWih,  eWih_b,  (long)2 * 2 * 2048 * 1024);
  cvt_bf16_k<<<2048, 256, 0, stream>>>(eWhh,  eWhh_b,  (long)2 * 2 * 2048 * 512);
  cvt_bf16_k<<<2048, 256, 0, stream>>>(dWih0, dWih0_b, (long)4096 * 2048);
  cvt_bf16_k<<<2048, 256, 0, stream>>>(dWih1, dWih1_b, (long)4096 * 1024);
  cvt_bf16_k<<<2048, 256, 0, stream>>>(dWhh,  dWhh_b,  (long)2 * 4096 * 1024);
  cvt_bf16_k<<<4096, 256, 0, stream>>>(Wout,  Wout_b,  (long)32000 * 1024);

  // 2. combined biases
  add2_k<<<(8192 + 255) / 256, 256, 0, stream>>>(eBih, eBhh, bcomb, 8192);
  add2_k<<<(8192 + 255) / 256, 256, 0, stream>>>(dBih, dBhh, dbcomb, 8192);

  // 3. embeddings
  embed_src_k<<<dim3(64, 32), 256, 0, stream>>>(x, src_emb, seq0);
  embed_tgt_k<<<dim3(63, 32), 256, 0, stream>>>(y, tgt_emb, inp0);

  // 4. encoder layers
  const uint16_t* seqin = seq0;
  for (int l = 0; l < 2; ++l) {
    (void)hipMemsetAsync(ehpool, 0, (size_t)2 * 2 * 32 * 512 * 2, stream);
    (void)hipMemsetAsync(ecpool, 0, (size_t)2 * 32 * 512 * 4, stream);
    for (int d = 0; d < 2; ++d) {
      gemm_k<false><<<dim3(16, 16), 256, 0, stream>>>(
          seqin, eWih_b + (size_t)(l * 2 + d) * 2048 * 1024,
          bcomb + (size_t)(l * 2 + d) * 2048,
          gp + (size_t)d * 2048 * 2048, 2048, 2048, 1024, 0, 0);
    }
    for (int t = 0; t < 64; ++t) {
      enc_step_k<<<dim3(4, 2), 256, 0, stream>>>(
          eWhh_b + (size_t)l * 2 * 2048 * 512, gp, ehpool, ecpool,
          (l == 0) ? seq1 : (uint16_t*)nullptr,
          hfin + (size_t)l * 32 * 1024, cfin + (size_t)l * 32 * 1024, t);
    }
    seqin = seq1;
  }

  // 5. decoder init + context broadcast + input-projection GEMM
  dec_init_k<<<128, 256, 0, stream>>>(hfin, cfin, h0pool, h1pool, c0, c1);
  ctx_fill_k<<<dim3(63, 32), 256, 0, stream>>>(hfin + (size_t)32 * 1024, inp0);
  gemm_k<false><<<dim3(32, 16), 256, 0, stream>>>(
      inp0, dWih0_b, dbcomb, P0, 2016, 4096, 2048, 0, 0);

  // 6. decoder recurrence
  for (int t = 0; t < 63; ++t) {
    dec_cell0_k<<<dim3(8), 256, 0, stream>>>(dWhh_b, P0, h0pool, c0, t);
    dec_cell1_k<<<dim3(8), 256, 0, stream>>>(
        dWih1_b, dWhh_b + (size_t)4096 * 1024, dbcomb + 4096,
        h0pool + (size_t)((t + 1) & 1) * 32 * 1024, h1pool, c1, h1all, t);
  }

  // 7. logits: h1_all @ W_out^T + b_out -> d_out with (t,b) -> (b,t) remap
  gemm_k<true><<<dim3(250, 16), 256, 0, stream>>>(
      h1all, Wout_b, bout, out, 2016, 32000, 1024,
      (long)63 * 32000, (long)32000);
}
